// NeuralTM_75797582840353
// MI455X (gfx1250) — compile-verified, matrix operand order checked by reading, not run on hardware
//
#include <hip/hip_runtime.h>

typedef __attribute__((ext_vector_type(16))) _Float16 v16h;
typedef __attribute__((ext_vector_type(8)))  _Float16 v8h;
typedef __attribute__((ext_vector_type(8)))  float    v8f;

namespace {

constexpr int T_STEPS = 8192;
constexpr int N_ROWS  = 1024;
constexpr int MT_LD   = 65;          // padded row stride (bank-conflict-free)

// GEMV tilings (16x16x32 f16 WMMA). N padded to multiples of 16.
constexpr int WC_KT  = 4,  WC_NT  = 32;  // Wc:      K=128, N=512
constexpr int WWR_KT = 16, WWR_NT = 18;  // Ww++Wr:  K=512, Npad=288
                                         //   cols [0,198)=Ww, [208,278)=Wr, rest 0
constexpr int WO_KT  = 2,  WO_NT  = 4;   // Wo:      K=64,  N=64 (bulk output GEMM)

constexpr int WC_F  = WC_KT  * WC_NT  * 512;  // 65536 halfs
constexpr int WWR_F = WWR_KT * WWR_NT * 512;  // 147456 halfs
constexpr int WO_F  = WO_KT  * WO_NT  * 512;  // 4096 halfs
constexpr int TOTAL_F = WC_F + WWR_F + WO_F;  // 217088 halfs = 424 KB

// LDS layout (float offsets); all buffers 16B-aligned.
// NOTE: OW and OR_ are contiguous: combined head-output buffer spans 416 floats,
// and the phase-C column index maps 1:1 onto it (cols 198..207 / 278..287 land
// in unused slack), enabling an unconditional store.
constexpr int S_MT  = 0;
constexpr int S_XHH = S_MT + N_ROWS * MT_LD;  // f16 x=[s;r] (128 halfs)
constexpr int S_HBH = S_XHH + 64;             // f16 h (512 halfs)
constexpr int S_OW  = S_HBH + 256;            // write-head pre-acts (208)
constexpr int S_OR  = S_OW + 208;             // read-head pre-acts (208)
constexpr int S_KV  = S_OR + 208;             // key (64)
constexpr int S_EV  = S_KV + 64;              // erase (64)
constexpr int S_AV  = S_EV + 64;              // add (64)
constexpr int S_WW  = S_AV + 64;              // write weights (1024)
constexpr int S_RW  = S_WW + 1024;            // read weights (1024)
constexpr int S_TMP = S_RW + 1024;            // wg scratch (1024)
constexpr int S_T2  = S_TMP + 1024;           // read partials (1024)
constexpr int S_RV  = S_T2 + 1024;            // r vector (64)
constexpr int S_RED = S_RV + 64;              // reduction tree, 2 slots (64)
constexpr int S_NRM = S_RED + 64;             // cached ||Mt row||^2 (1024)
constexpr int SMEM_FLOATS = S_NRM + 1024;     // ~284 KB < 320 KB

// Workgroup barrier WITHOUT the global-store fence: history writes to d_out are
// fire-and-forget (never re-read in-kernel), so only DS ordering is required.
__device__ __forceinline__ void wg_sync() {
  asm volatile("s_wait_dscnt 0x0\n\t"
               "s_barrier_signal -1\n\t"
               "s_barrier_wait -1" ::: "memory");
}

// ---- branch-free fast math (v_exp/v_log/v_rcp/v_sqrt) ----
__device__ __forceinline__ float fast_div(float a, float b) {
  return a * __builtin_amdgcn_rcpf(b);           // v_rcp_f32 + v_mul
}
__device__ __forceinline__ float fast_sigmoid(float x) {
  return __builtin_amdgcn_rcpf(1.f + __expf(-x));
}
__device__ __forceinline__ float fast_tanh(float x) {
  x = fminf(fmaxf(x, -15.f), 15.f);
  float e = __expf(2.f * x);
  return fast_div(e - 1.f, e + 1.f);
}
__device__ __forceinline__ float fast_softplus(float x) {
  return fmaxf(x, 0.f) + __logf(1.f + __expf(-fabsf(x)));
}
// x >= 0 only; pow(0,g) -> exp(-inf) = 0 (correct)
__device__ __forceinline__ float fast_pow_nn(float x, float y) {
  return __expf(y * __logf(x));
}

// Broadcast A fragment: activation vector sits in matrix row M=0, and since we
// only ever read D row 0, rows 1..15 may hold anything -> ALL lanes load the
// row-0 pattern (no exec divergence). 16-bit A 16x32 layout:
//   K = (t<8 ? t : t+8) + 8*(lane>=16)  =>  halfs [k0+koff, +8) and [k0+koff+16, +8)
__device__ __forceinline__ v16h loadA_h(const _Float16* v, int k0, int lane) {
  int koff = (lane >> 4) << 3;
  v8h lo = *(const v8h*)(v + k0 + koff);        // ds_load_b128
  v8h hi = *(const v8h*)(v + k0 + koff + 16);   // ds_load_b128
  return __builtin_shufflevector(lo, hi, 0, 1, 2, 3, 4, 5, 6, 7,
                                 8, 9, 10, 11, 12, 13, 14, 15);
}

// Double-buffered block reduction: consecutive calls must alternate `slot`,
// which makes the trailing barrier unnecessary.
__device__ __forceinline__ float block_reduce(float v, bool do_sum, float* red,
                                              int slot, int tid) {
  float* r = red + (slot << 5);
#pragma unroll
  for (int m = 16; m > 0; m >>= 1) {
    float o = __shfl_xor(v, m, 32);
    v = do_sum ? (v + o) : fmaxf(v, o);
  }
  if ((tid & 31) == 0) r[tid >> 5] = v;
  wg_sync();
  if (tid < 32) {
    float w = r[tid];
#pragma unroll
    for (int m = 16; m > 0; m >>= 1) {
      float o = __shfl_xor(w, m, 32);
      w = do_sum ? (w + o) : fmaxf(w, o);
    }
    if (tid == 0) r[0] = w;
  }
  wg_sync();
  return r[0];
}

// Full NTM addressing: cosine sim -> softmax -> interpolate -> shift -> sharpen.
// Updates W[tid] in place (and returns it). One memory row per thread.
// Row norms come from the maintained NRM cache (updated in the memory-update
// phase), saving a full recompute pass per call.
__device__ float ntm_address(const float* MT, const float* KV, const float* NRM,
                             float beta, float g,
                             float sh0, float sh1, float sh2, float gamma,
                             float* W, float* TMP, float* RED, int slot0, int tid) {
  float dot = 0.f, knrm2 = 0.f;
  const float* row = MT + tid * MT_LD;
#pragma unroll 8
  for (int j = 0; j < 64; ++j) {
    float m = row[j];
    float kk = KV[j];              // LDS broadcast, shared with knrm2
    dot += m * kk;
    knrm2 += kk * kk;
  }
  float nrm = NRM[tid];
  float sim = fast_div(dot, __builtin_amdgcn_sqrtf(nrm) *
                            __builtin_amdgcn_sqrtf(knrm2) + 1e-16f);
  float z = beta * sim;
  float mx = block_reduce(z, false, RED, slot0, tid);
  float ez = __expf(z - mx);
  float se = block_reduce(ez, true, RED, slot0 ^ 1, tid);
  float wc = fast_div(ez, se);
  float wg = g * wc + (1.f - g) * W[tid];
  TMP[tid] = wg;
  wg_sync();
  float ws = sh0 * TMP[(tid + 1023) & 1023] + sh1 * TMP[tid] + sh2 * TMP[(tid + 1) & 1023];
  float wsh = fast_pow_nn(ws, gamma);
  float ssum = block_reduce(wsh, true, RED, slot0, tid);
  float wn = fast_div(wsh, ssum + 1e-16f);
  W[tid] = wn;
  wg_sync();
  return wn;
}

// Pack f32 weights into WMMA-fragment-native f16 tiles (B-operand order):
// one tile = 32 lanes x 16 halfs contiguous, so each lane loads 2x b128.
// Region 0: Wc (32 n-tiles). Region 1: Ww|pad|Wr (18 n-tiles). Region 2: Wo.
__global__ void ntm_pack_weights(const float* __restrict__ Wc,
                                 const float* __restrict__ Ww,
                                 const float* __restrict__ Wr,
                                 const float* __restrict__ Wo,
                                 _Float16* __restrict__ frag) {
  int g = blockIdx.x * blockDim.x + threadIdx.x;
  if (g >= TOTAL_F) return;
  int e = g & 511;       // element within tile
  int q = g >> 9;        // tile index across regions
  int lane = e >> 4;
  int tt = e & 15;
  int n_loc = lane & 15;
  int k_loc = ((tt < 8) ? tt : tt + 8) + ((lane >> 4) << 3);
  float val = 0.f;
  if (q < WC_KT * WC_NT) {
    int nt = q / WC_KT, kt = q % WC_KT;
    val = Wc[(kt * 32 + k_loc) * 512 + nt * 16 + n_loc];
  } else if (q < WC_KT * WC_NT + WWR_KT * WWR_NT) {
    int qq = q - WC_KT * WC_NT;
    int nt = qq / WWR_KT, kt = qq % WWR_KT;
    int n = nt * 16 + n_loc;
    int k = kt * 32 + k_loc;
    if (n < 198)                    val = Ww[k * 198 + n];
    else if (n >= 208 && n < 278)   val = Wr[k * 70 + (n - 208)];
  } else {
    int qq = q - WC_KT * WC_NT - WWR_KT * WWR_NT;
    int nt = qq / WO_KT, kt = qq % WO_KT;
    val = Wo[(kt * 32 + k_loc) * 64 + nt * 16 + n_loc];
  }
  frag[g] = (_Float16)val;
}

__global__ void __launch_bounds__(1024, 1)
ntm_persistent(const float* __restrict__ seq,
               const float* __restrict__ bc,
               const float* __restrict__ br,
               const float* __restrict__ bw,
               const _Float16* __restrict__ fWc,
               const _Float16* __restrict__ fWwr,
               float* __restrict__ out) {
  extern __shared__ float sm[];
  const int tid  = threadIdx.x;
  const int wave = tid >> 5;
  const int lane = tid & 31;

  float*     MT  = sm + S_MT;
  _Float16*  XHh = (_Float16*)(sm + S_XHH);
  _Float16*  HBh = (_Float16*)(sm + S_HBH);
  float*     OW  = sm + S_OW;   // 416-float combined [ow | orr] region
  float*     OR_ = sm + S_OR;
  float*     KV  = sm + S_KV;
  float*     EV  = sm + S_EV;
  float*     AV  = sm + S_AV;
  float*     WW  = sm + S_WW;
  float*     RW  = sm + S_RW;
  float*     TMP = sm + S_TMP;
  float*     T2  = sm + S_T2;
  float*     RV  = sm + S_RV;
  float*     RED = sm + S_RED;
  float*     NRM = sm + S_NRM;

  float* out_ww  = out + (size_t)T_STEPS * 64;
  float* out_rw  = out_ww + (size_t)T_STEPS * 1024;
  float* out_a   = out_rw + (size_t)T_STEPS * 1024;
  float* out_r   = out_a + (size_t)T_STEPS * 64;

  // ---- loop-invariant per-wave bias scalars (keep global loads out of t-loop)
  const int bidx = wave * 16 + (lane & 15);
  const float bc_r = bc[bidx];                       // bidx < 512 always
  float bwr_r = 0.f;
  if (wave < WWR_NT) {
    if (bidx < 198)                     bwr_r = bw[bidx];
    else if (bidx >= 208 && bidx < 278) bwr_r = br[bidx - 208];
  }

  // ---- hoist Wc B-fragments into REGISTERS (immune to the wg_sync memory
  // clobber, unlike re-loadable global memory): 4 x v16h = 32 VGPRs.
  const v16h* BCp = (const v16h*)fWc + (size_t)wave * (WC_KT * 32) + lane;
  const v16h wcb0 = BCp[0];
  const v16h wcb1 = BCp[32];
  const v16h wcb2 = BCp[64];
  const v16h wcb3 = BCp[96];

  const v16h* BWRp = (const v16h*)fWwr + (size_t)wave * (WWR_KT * 32) + lane;

  // ---- initial carry state ----
  {
    float* row = MT + tid * MT_LD;
#pragma unroll 8
    for (int j = 0; j < 64; ++j) row[j] = 1e-6f;
    row[64] = 0.f;
    WW[tid] = (tid == 512) ? 1.f : 0.f;
    RW[tid] = (tid == 512) ? 1.f : 0.f;
    NRM[tid] = 64.f * 1e-12f;              // ||init row||^2
    if (tid < 64) RV[tid] = 1e-6f;
  }
  float s_reg = (tid < 64) ? seq[tid] : 0.f;   // software prefetch of s_0
  __syncthreads();

  for (int t = 0; t < T_STEPS; ++t) {
    // A: x = [s_t ; r_prev] (f16; only WMMA consumes it), then prefetch s_{t+1}
    if (tid < 64)       XHh[tid] = (_Float16)s_reg;
    else if (tid < 128) XHh[tid] = (_Float16)RV[tid - 64];
    if (tid < 64 && t + 1 < T_STEPS) s_reg = seq[(t + 1) * 64 + tid];
    wg_sync();

    // Prefetch phase-C's first B group NOW: overlaps the phase-B WMMAs and the
    // barrier, hiding one L2 round-trip.
    v16h cb0, cb1;
    if (wave < WWR_NT) { cb0 = BWRp[0]; cb1 = BWRp[32]; }

    // B: h = tanh(x @ Wc + bc) — 32 waves, one 16-col tile each; B operands
    // live entirely in registers, so this phase is pure LDS + WMMA.
    {
      v16h a0 = loadA_h(XHh, 0,  lane);
      v16h a1 = loadA_h(XHh, 32, lane);
      v16h a2 = loadA_h(XHh, 64, lane);
      v16h a3 = loadA_h(XHh, 96, lane);
      v8f acc = {};
      acc = __builtin_amdgcn_wmma_f32_16x16x32_f16(false, a0, false, wcb0,
                                                   (short)0, acc, false, false);
      acc = __builtin_amdgcn_wmma_f32_16x16x32_f16(false, a1, false, wcb1,
                                                   (short)0, acc, false, false);
      acc = __builtin_amdgcn_wmma_f32_16x16x32_f16(false, a2, false, wcb2,
                                                   (short)0, acc, false, false);
      acc = __builtin_amdgcn_wmma_f32_16x16x32_f16(false, a3, false, wcb3,
                                                   (short)0, acc, false, false);
      if (lane < 16) {
        int idx = wave * 16 + lane;
        HBh[idx] = (_Float16)fast_tanh(acc[0] + bc_r);  // D row 0, lanes 0..15
      }
    }
    wg_sync();

    // C: [ow | orr] = h @ [Ww | Wr] — waves 0..17, one 16-col tile each.
    // Software-pipelined: group k+1's B loads issue while group k's WMMAs run,
    // so only ~one L2 latency is exposed across all 8 groups.
    if (wave < WWR_NT) {
      v16h b0 = cb0, b1 = cb1;
      v8f acc = {};
#pragma unroll
      for (int kb = 0; kb < 8; ++kb) {
        v16h a0 = loadA_h(HBh, (kb * 2 + 0) * 32, lane);
        v16h a1 = loadA_h(HBh, (kb * 2 + 1) * 32, lane);
        v16h nb0 = b0, nb1 = b1;
        if (kb < 7) {
          nb0 = BWRp[(kb * 2 + 2) * 32];
          nb1 = BWRp[(kb * 2 + 3) * 32];
        }
        acc = __builtin_amdgcn_wmma_f32_16x16x32_f16(false, a0, false, b0,
                                                     (short)0, acc, false, false);
        acc = __builtin_amdgcn_wmma_f32_16x16x32_f16(false, a1, false, b1,
                                                     (short)0, acc, false, false);
        b0 = nb0; b1 = nb1;
      }
      // Combined [ow|orr] LDS region: column index maps 1:1, pad cols land in
      // unused slack -> single unconditional store, no compares.
      if (lane < 16) OW[wave * 16 + lane] = acc[0] + bwr_r;
    }
    wg_sync();

    // D: write-head vectors
    if (tid < 64)       KV[tid] = fast_tanh(OW[tid]);
    else if (tid < 128) EV[tid - 64]  = fast_sigmoid(OW[70 + (tid - 64)]);
    else if (tid < 192) AV[tid - 128] = fast_tanh(OW[134 + (tid - 128)]);
    wg_sync();

    // E: write addressing (scalars recomputed per-thread from LDS broadcasts)
    {
      float beta  = fast_softplus(OW[64]);
      float g     = fast_sigmoid(OW[65]);
      float s0 = OW[66], s1 = OW[67], s2 = OW[68];
      float mx = fmaxf(s0, fmaxf(s1, s2));
      float e0 = __expf(s0 - mx), e1 = __expf(s1 - mx), e2 = __expf(s2 - mx);
      float ri = __builtin_amdgcn_rcpf(e0 + e1 + e2);
      float gamma = 1.f + fast_softplus(OW[69]);
      float wn = ntm_address(MT, KV, NRM, beta, g, e0 * ri, e1 * ri, e2 * ri,
                             gamma, WW, TMP, RED, /*slot0=*/0, tid);
      out_ww[(size_t)t * 1024 + tid] = wn;
    }

    // F: memory update, one row per thread; refresh the row-norm cache for free
    {
      float wwi = WW[tid];
      float* row = MT + tid * MT_LD;
      float nn = 0.f;
#pragma unroll 8
      for (int j = 0; j < 64; ++j) {
        float v = row[j] * (1.f - wwi * EV[j]) + wwi * AV[j];
        row[j] = v;
        nn += v * v;
      }
      NRM[tid] = nn;
      if (tid < 64) out_a[(size_t)t * 64 + tid] = AV[tid];
    }
    wg_sync();

    // G: read head
    if (tid < 64) KV[tid] = fast_tanh(OR_[tid]);
    wg_sync();
    {
      float beta  = fast_softplus(OR_[64]);
      float g     = fast_sigmoid(OR_[65]);
      float s0 = OR_[66], s1 = OR_[67], s2 = OR_[68];
      float mx = fmaxf(s0, fmaxf(s1, s2));
      float e0 = __expf(s0 - mx), e1 = __expf(s1 - mx), e2 = __expf(s2 - mx);
      float ri = __builtin_amdgcn_rcpf(e0 + e1 + e2);
      float gamma = 1.f + fast_softplus(OR_[69]);
      float wn = ntm_address(MT, KV, NRM, beta, g, e0 * ri, e1 * ri, e2 * ri,
                             gamma, RW, TMP, RED, /*slot0=*/1, tid);
      out_rw[(size_t)t * 1024 + tid] = wn;
    }

    // H: r = rw @ Mt  (16 partial groups x 64 cols)
    {
      int j = tid & 63, grp = tid >> 6;
      float p = 0.f;
      const float* base = MT + (grp * 64) * MT_LD + j;
#pragma unroll 8
      for (int ii = 0; ii < 64; ++ii) p += RW[grp * 64 + ii] * base[ii * MT_LD];
      T2[grp * 64 + j] = p;
    }
    wg_sync();
    if (tid < 64) {
      float s = 0.f;
#pragma unroll
      for (int gI = 0; gI < 16; ++gI) s += T2[gI * 64 + tid];
      RV[tid] = s;
      out_r[(size_t)t * 64 + tid] = s;
    }
    wg_sync();
    // (output GEMV moved out of the recurrence into ntm_output)
  }
}

// Bulk output projection: seq_out = clip(r_h @ Wo + bo, 0, 1).
// [8192x64]@[64x64] -> 512 independent 16-row WMMA tiles (one per wave).
__global__ void __launch_bounds__(256)
ntm_output(const float* __restrict__ r_h,
           const float* __restrict__ bo,
           const _Float16* __restrict__ fWo,
           float* __restrict__ out_seq) {
  int gwave = (blockIdx.x * blockDim.x + threadIdx.x) >> 5;  // tile 0..511
  int lane  = threadIdx.x & 31;
  if (gwave >= T_STEPS / 16) return;
  int row0 = gwave * 16;
  int m    = lane & 15;
  int koff = (lane >> 4) << 3;
  const float* rrow = r_h + (size_t)(row0 + m) * 64;

  v16h a0, a1;
#pragma unroll
  for (int tt2 = 0; tt2 < 8; ++tt2) {
    a0[tt2]     = (_Float16)rrow[koff + tt2];
    a0[tt2 + 8] = (_Float16)rrow[koff + 16 + tt2];
    a1[tt2]     = (_Float16)rrow[32 + koff + tt2];
    a1[tt2 + 8] = (_Float16)rrow[32 + koff + 16 + tt2];
  }
  const v16h* B = (const v16h*)fWo + lane;
#pragma unroll
  for (int nt = 0; nt < WO_NT; ++nt) {
    v8f acc = {};
    acc = __builtin_amdgcn_wmma_f32_16x16x32_f16(false, a0, false, B[(nt * 2 + 0) * 32],
                                                 (short)0, acc, false, false);
    acc = __builtin_amdgcn_wmma_f32_16x16x32_f16(false, a1, false, B[(nt * 2 + 1) * 32],
                                                 (short)0, acc, false, false);
    int col  = nt * 16 + (lane & 15);
    float bb = bo[col];
    int rbase = row0 + ((lane < 16) ? 0 : 8);  // D: VGPR r -> M=r (+8 for hi lanes)
#pragma unroll
    for (int rr = 0; rr < 8; ++rr) {
      float v = acc[rr] + bb;
      out_seq[(size_t)(rbase + rr) * 64 + col] = fminf(fmaxf(v, 0.f), 1.f);
    }
  }
}

} // namespace

extern "C" void kernel_launch(void* const* d_in, const int* in_sizes, int n_in,
                              void* d_out, int out_size, void* d_ws, size_t ws_size,
                              hipStream_t stream) {
  const float* seq = (const float*)d_in[0];
  const float* Wc  = (const float*)d_in[1];
  const float* bc  = (const float*)d_in[2];
  const float* Wr  = (const float*)d_in[3];
  const float* br  = (const float*)d_in[4];
  const float* Ww  = (const float*)d_in[5];
  const float* bw  = (const float*)d_in[6];
  const float* Wo  = (const float*)d_in[7];
  const float* bo  = (const float*)d_in[8];

  _Float16* frag = (_Float16*)d_ws;            // 424 KB of scratch used
  _Float16* fWc  = frag;
  _Float16* fWwr = fWc + WC_F;
  _Float16* fWo  = fWwr + WWR_F;

  float* out   = (float*)d_out;
  float* out_r = out + (size_t)T_STEPS * (64 + 1024 + 1024 + 64);  // r_h region

  ntm_pack_weights<<<(TOTAL_F + 255) / 256, 256, 0, stream>>>(Wc, Ww, Wr, Wo, frag);

  size_t smem = (size_t)SMEM_FLOATS * sizeof(float);   // ~284 KB LDS (<320 KB/WGP)
  ntm_persistent<<<1, 1024, smem, stream>>>(seq, bc, br, bw, fWc, fWwr, out);

  ntm_output<<<(T_STEPS / 16 * 32 + 255) / 256, 256, 0, stream>>>(out_r, bo, fWo, out);
}
